// RNN_34187939676743
// MI455X (gfx1250) — matrix operand order repaired and checked by reading, not compiled, not run on hardware
//
#include <hip/hip_runtime.h>
#include <hip/hip_bf16.h>

// ---------------- problem constants ----------------
#define N_TOT   2048
#define SEQ     70
#define HID     230
#define GATES   690          // 3*HID
#define NPAD    704          // GATES padded to mult of 32
#define KPAD_H  256          // HID padded to mult of 32
#define KT_H    8            // KPAD_H/32
#define D_IN    60
#define KPAD_X  64
#define KT_X    2
#define EMB_D   50
#define POS_D   5
#define NBAG    64
#define NREL    100
#define HSTR    232          // padded hidden stride for tup/repre

typedef __attribute__((ext_vector_type(16))) __bf16 v16bf;
typedef __attribute__((ext_vector_type(8)))  float  v8f;

union Frag16 { v16bf v; uint4 q[2]; };

// ---------------- WMMA GEMM: D[M x NPAD](f32) = A[M x 32*ktiles](bf16) @ Bp ----------------
// A: row-major, leading dim lda (elements). Bp: packed [kt][NPAD][32] bf16
// where Bp[kt][n][kl] = B[kt*32+kl][n]. One wave computes a 32x32 tile.
__global__ void __launch_bounds__(32)
wmma_gemm_bf16_kernel(const __hip_bfloat16* __restrict__ A, int lda,
                      const __hip_bfloat16* __restrict__ Bp,
                      float* __restrict__ D, int ldd, int ktiles)
{
    const int lane = threadIdx.x;        // 0..31
    const int m0   = blockIdx.x * 32;
    const int n0   = blockIdx.y * 32;
    const int r    = lane & 15;          // row/col within 16-tile
    const int kh   = lane >> 4;          // lane-half selects K sub-range

    v8f acc00 = {}; v8f acc01 = {}; v8f acc10 = {}; v8f acc11 = {};

    for (int kt = 0; kt < ktiles; ++kt) {
        Frag16 a0, a1, b0, b1;
        // A fragment: lane holds row m0(+16)+r; elements 0..7 -> k = kt*32+kh*8+e,
        // elements 8..15 -> k = kt*32+16+kh*8+(e-8)  (ISA 16-bit A 16x32 layout)
        const __hip_bfloat16* ap0 = A + (size_t)(m0 + r) * lda + kt * 32 + kh * 8;
        const __hip_bfloat16* ap1 = ap0 + (size_t)16 * lda;
        a0.q[0] = *(const uint4*)(ap0);
        a0.q[1] = *(const uint4*)(ap0 + 16);
        a1.q[0] = *(const uint4*)(ap1);
        a1.q[1] = *(const uint4*)(ap1 + 16);
        // B fragment: lane holds col n0(+16)+r; elements 0..7 -> k_local = kh*8+e, 8..15 -> 16+kh*8+(e-8)
        const __hip_bfloat16* bp0 = Bp + ((size_t)kt * NPAD + (n0 + r)) * 32 + kh * 8;
        const __hip_bfloat16* bp1 = bp0 + 16 * 32;
        b0.q[0] = *(const uint4*)(bp0);
        b0.q[1] = *(const uint4*)(bp0 + 16);
        b1.q[0] = *(const uint4*)(bp1);
        b1.q[1] = *(const uint4*)(bp1 + 16);

        acc00 = __builtin_amdgcn_wmma_f32_16x16x32_bf16(false, a0.v, false, b0.v, (short)0, acc00, false, false);
        acc01 = __builtin_amdgcn_wmma_f32_16x16x32_bf16(false, a0.v, false, b1.v, (short)0, acc01, false, false);
        acc10 = __builtin_amdgcn_wmma_f32_16x16x32_bf16(false, a1.v, false, b0.v, (short)0, acc10, false, false);
        acc11 = __builtin_amdgcn_wmma_f32_16x16x32_bf16(false, a1.v, false, b1.v, (short)0, acc11, false, false);
    }

    // C/D layout: VGPR v -> row m0 + kh*8 + v (+16 for second tile), col n0 + r (+16)
#pragma unroll
    for (int v = 0; v < 8; ++v) {
        const int row0 = m0 + kh * 8 + v;
        const int row1 = row0 + 16;
        const int col0 = n0 + r;
        const int col1 = col0 + 16;
        D[(size_t)row0 * ldd + col0] = acc00[v];
        D[(size_t)row0 * ldd + col1] = acc01[v];
        D[(size_t)row1 * ldd + col0] = acc10[v];
        D[(size_t)row1 * ldd + col1] = acc11[v];
    }
}

// ---------------- embedding gather -> padded bf16 [N*T][64] ----------------
__global__ void embed_kernel(const int* __restrict__ sentence, const int* __restrict__ pos1,
                             const int* __restrict__ pos2, const float* __restrict__ word_emb,
                             const float* __restrict__ p1t, const float* __restrict__ p2t,
                             __hip_bfloat16* __restrict__ embPad)
{
    long idx = (long)blockIdx.x * blockDim.x + threadIdx.x;
    if (idx >= (long)N_TOT * SEQ * KPAD_X) return;
    const long row = idx >> 6;
    const int  c   = (int)(idx & 63);
    float v = 0.f;
    if (c < EMB_D)                  v = word_emb[(size_t)sentence[row] * EMB_D + c];
    else if (c < EMB_D + POS_D)     v = p1t[(size_t)pos1[row] * POS_D + (c - EMB_D)];
    else if (c < D_IN)              v = p2t[(size_t)pos2[row] * POS_D + (c - EMB_D - POS_D)];
    embPad[idx] = __float2bfloat16(v);
}

// ---------------- weight packing into B-fragment layout ----------------
// Bp[kt][n][kl] = (n<GATES && kt*32+kl<Ksrc) ? W[n*Ksrc + kt*32+kl] : 0
__global__ void pack_w_kernel(const float* __restrict__ W, __hip_bfloat16* __restrict__ Bp,
                              int ktiles, int Ksrc)
{
    long idx = (long)blockIdx.x * blockDim.x + threadIdx.x;
    const long total = (long)ktiles * NPAD * 32;
    if (idx >= total) return;
    const int kt  = (int)(idx / (NPAD * 32));
    const int rem = (int)(idx - (long)kt * NPAD * 32);
    const int n   = rem >> 5;
    const int kl  = rem & 31;
    const int k   = kt * 32 + kl;
    float v = (n < GATES && k < Ksrc) ? W[(size_t)n * Ksrc + k] : 0.f;
    Bp[idx] = __float2bfloat16(v);
}

__global__ void zero_f32_kernel(float* __restrict__ p, long n)
{
    long i = (long)blockIdx.x * blockDim.x + threadIdx.x;
    if (i < n) p[i] = 0.f;
}

// ---------------- GRU gate update for one direction, one timestep ----------------
__global__ void gru_gate_kernel(const float* __restrict__ gh, const float* __restrict__ gi,
                                const float* __restrict__ b_ih, const float* __restrict__ b_hh,
                                float* __restrict__ hf32, __hip_bfloat16* __restrict__ hbf,
                                float* __restrict__ tup, int t)
{
    int idx = blockIdx.x * blockDim.x + threadIdx.x;
    if (idx >= N_TOT * HID) return;
    const int n = idx / HID;
    const int j = idx - n * HID;
    const float* gin = gi + (size_t)n * NPAD;
    const float* ghn = gh + (size_t)n * NPAD;
    const float ir  = gin[j]           + b_ih[j];
    const float iz  = gin[HID + j]     + b_ih[HID + j];
    const float inn = gin[2 * HID + j] + b_ih[2 * HID + j];
    const float hr  = ghn[j]           + b_hh[j];
    const float hz  = ghn[HID + j]     + b_hh[HID + j];
    const float hn  = ghn[2 * HID + j] + b_hh[2 * HID + j];
    const float rg = 1.f / (1.f + expf(-(ir + hr)));
    const float zg = 1.f / (1.f + expf(-(iz + hz)));
    const float ng = tanhf(inn + rg * hn);
    const float hp = hf32[(size_t)n * KPAD_H + j];
    const float h  = (1.f - zg) * ng + zg * hp;
    hf32[(size_t)n * KPAD_H + j] = h;
    hbf [(size_t)n * KPAD_H + j] = __float2bfloat16(h);
    tup[((size_t)n * SEQ + t) * HSTR + j] += h;
}

// ---------------- word-level attention over T, per sentence ----------------
__global__ void word_attn_kernel(const float* __restrict__ tup,
                                 const float* __restrict__ attention_w,
                                 float* __restrict__ repre)
{
    const int n   = blockIdx.x;
    const int tid = threadIdx.x;          // 256
    const float* tp = tup + (size_t)n * SEQ * HSTR;
    __shared__ float sc[SEQ];
    __shared__ float al[SEQ];
    if (tid < SEQ) sc[tid] = 0.f;
    __syncthreads();
    for (int idx = tid; idx < SEQ * HID; idx += 256) {
        const int t = idx / HID;
        const int j = idx - t * HID;
        atomicAdd(&sc[t], tanhf(tp[(size_t)t * HSTR + j]) * attention_w[j]);
    }
    __syncthreads();
    if (tid == 0) {
        float m = -1e30f;
        for (int t = 0; t < SEQ; ++t) m = fmaxf(m, sc[t]);
        float s = 0.f;
        for (int t = 0; t < SEQ; ++t) { float e = expf(sc[t] - m); al[t] = e; s += e; }
        const float inv = 1.f / s;
        for (int t = 0; t < SEQ; ++t) al[t] *= inv;
    }
    __syncthreads();
    for (int j = tid; j < HID; j += 256) {
        float acc = 0.f;
        for (int t = 0; t < SEQ; ++t) acc += al[t] * tp[(size_t)t * HSTR + j];
        repre[(size_t)n * HSTR + j] = tanhf(acc);
    }
}

// ---------------- ragged bag attention (segment softmax) ----------------
__global__ void bag_attn_kernel(const float* __restrict__ repre,
                                const float* __restrict__ sen_a,
                                const float* __restrict__ sen_r,
                                const int* __restrict__ total_shape,
                                float* __restrict__ sen_s)
{
    const int b   = blockIdx.x;
    const int tid = threadIdx.x;          // 256
    const int lo  = total_shape[b];
    const int hi  = total_shape[b + 1];
    int cnt = hi - lo;
    if (cnt > 128) cnt = 128;             // shared-memory bound (actual: 32)
    __shared__ float sv[128];
    __shared__ float al[128];
    if (tid < 128) sv[tid] = 0.f;
    __syncthreads();
    for (int idx = tid; idx < cnt * HID; idx += 256) {
        const int i = idx / HID;
        const int j = idx - i * HID;
        atomicAdd(&sv[i], repre[(size_t)(lo + i) * HSTR + j] * sen_a[j] * sen_r[j]);
    }
    __syncthreads();
    if (tid == 0) {
        float m = -1e30f;
        for (int i = 0; i < cnt; ++i) m = fmaxf(m, sv[i]);
        float s = 0.f;
        for (int i = 0; i < cnt; ++i) { float e = expf(sv[i] - m); al[i] = e; s += e; }
        const float inv = 1.f / s;
        for (int i = 0; i < cnt; ++i) al[i] *= inv;
    }
    __syncthreads();
    for (int j = tid; j < HID; j += 256) {
        float acc = 0.f;
        for (int i = 0; i < cnt; ++i) acc += al[i] * repre[(size_t)(lo + i) * HSTR + j];
        sen_s[(size_t)b * HSTR + j] = acc;
    }
}

// ---------------- logits, softmax prob, BCE loss, accuracy ----------------
// out layout: [0]=total_loss, [1..64]=acc, [65..6464]=prob (64x100)
__global__ void logits_loss_kernel(const float* __restrict__ sen_s,
                                   const float* __restrict__ rel,
                                   const float* __restrict__ sen_d,
                                   const float* __restrict__ y,
                                   float* __restrict__ out)
{
    const int b   = blockIdx.x;
    const int tid = threadIdx.x;          // 128
    __shared__ float lg[NREL];
    for (int c = tid; c < NREL; c += 128) {
        float acc = sen_d[c];
        const float* ss = sen_s + (size_t)b * HSTR;
        const float* rr = rel + (size_t)c * HID;
        for (int j = 0; j < HID; ++j) acc += ss[j] * rr[j];
        lg[c] = acc;
    }
    __syncthreads();
    if (tid == 0) {
        float m = -1e30f; int am = 0;
        for (int c = 0; c < NREL; ++c) if (lg[c] > m) { m = lg[c]; am = c; }
        float s = 0.f;
        for (int c = 0; c < NREL; ++c) s += expf(lg[c] - m);
        const float inv = 1.f / s;
        float ym = -1e30f; int ya = 0;
        for (int c = 0; c < NREL; ++c) { float yv = y[(size_t)b * NREL + c]; if (yv > ym) { ym = yv; ya = c; } }
        out[1 + b] = (am == ya) ? 1.f : 0.f;
        float bce = 0.f;
        for (int c = 0; c < NREL; ++c) {
            const float x = lg[c];
            bce += fmaxf(x, 0.f) - x * y[(size_t)b * NREL + c] + log1pf(expf(-fabsf(x)));
            out[65 + b * NREL + c] = expf(x - m) * inv;
        }
        atomicAdd(&out[0], bce / (float)NREL);
    }
}

// ---------------- host orchestration ----------------
extern "C" void kernel_launch(void* const* d_in, const int* in_sizes, int n_in,
                              void* d_out, int out_size, void* d_ws, size_t ws_size,
                              hipStream_t stream)
{
    (void)in_sizes; (void)n_in; (void)out_size; (void)ws_size;
    const int*   sentence  = (const int*)d_in[0];
    const int*   pos1      = (const int*)d_in[1];
    const int*   pos2      = (const int*)d_in[2];
    const int*   tshape    = (const int*)d_in[3];
    const float* y_batch   = (const float*)d_in[4];
    const float* word_emb  = (const float*)d_in[5];
    const float* p1t       = (const float*)d_in[6];
    const float* p2t       = (const float*)d_in[7];
    const float* W_ih_f    = (const float*)d_in[8];
    const float* W_hh_f    = (const float*)d_in[9];
    const float* b_ih_f    = (const float*)d_in[10];
    const float* b_hh_f    = (const float*)d_in[11];
    const float* W_ih_b    = (const float*)d_in[12];
    const float* W_hh_b    = (const float*)d_in[13];
    const float* b_ih_b    = (const float*)d_in[14];
    const float* b_hh_b    = (const float*)d_in[15];
    const float* attn_w    = (const float*)d_in[16];
    const float* sen_a     = (const float*)d_in[17];
    const float* sen_r     = (const float*)d_in[18];
    const float* rel_emb   = (const float*)d_in[19];
    const float* sen_d     = (const float*)d_in[20];
    float* out = (float*)d_out;

    // -------- workspace carve (256B aligned) --------
    char* ws = (char*)d_ws;
    size_t off = 0;
#define TAKE(T, NAME, BYTES) T NAME = (T)(ws + off); off += (((size_t)(BYTES)) + 255) & ~(size_t)255
    TAKE(__hip_bfloat16*, embPad, (size_t)N_TOT * SEQ * KPAD_X * 2);
    TAKE(__hip_bfloat16*, BpF_hh, (size_t)KT_H * NPAD * 32 * 2);
    TAKE(__hip_bfloat16*, BpB_hh, (size_t)KT_H * NPAD * 32 * 2);
    TAKE(__hip_bfloat16*, BpF_ih, (size_t)KT_X * NPAD * 32 * 2);
    TAKE(__hip_bfloat16*, BpB_ih, (size_t)KT_X * NPAD * 32 * 2);
    TAKE(__hip_bfloat16*, hbf_f,  (size_t)N_TOT * KPAD_H * 2);
    TAKE(__hip_bfloat16*, hbf_b,  (size_t)N_TOT * KPAD_H * 2);
    TAKE(float*,          hf_f,   (size_t)N_TOT * KPAD_H * 4);
    TAKE(float*,          hf_b,   (size_t)N_TOT * KPAD_H * 4);
    TAKE(float*,          gh_f,   (size_t)N_TOT * NPAD * 4);
    TAKE(float*,          gh_b,   (size_t)N_TOT * NPAD * 4);
    TAKE(float*,          gi_f,   (size_t)N_TOT * NPAD * 4);
    TAKE(float*,          gi_b,   (size_t)N_TOT * NPAD * 4);
    TAKE(float*,          tup,    (size_t)N_TOT * SEQ * HSTR * 4);
    TAKE(float*,          repre,  (size_t)N_TOT * HSTR * 4);
    TAKE(float*,          sen_s,  (size_t)NBAG * HSTR * 4);
#undef TAKE

    // -------- init: zero h state (incl. bf16 pads), tup, output --------
    {
        long n;
        n = (long)N_TOT * KPAD_H / 2; // bf16 buffer zeroed as floats
        zero_f32_kernel<<<(unsigned)((n + 255) / 256), 256, 0, stream>>>((float*)hbf_f, n);
        zero_f32_kernel<<<(unsigned)((n + 255) / 256), 256, 0, stream>>>((float*)hbf_b, n);
        n = (long)N_TOT * KPAD_H;
        zero_f32_kernel<<<(unsigned)((n + 255) / 256), 256, 0, stream>>>(hf_f, n);
        zero_f32_kernel<<<(unsigned)((n + 255) / 256), 256, 0, stream>>>(hf_b, n);
        n = (long)N_TOT * SEQ * HSTR;
        zero_f32_kernel<<<(unsigned)((n + 255) / 256), 256, 0, stream>>>(tup, n);
        n = 1 + NBAG + NBAG * NREL;
        zero_f32_kernel<<<(unsigned)((n + 255) / 256), 256, 0, stream>>>(out, n);
    }

    // -------- embedding + weight packing --------
    {
        long n = (long)N_TOT * SEQ * KPAD_X;
        embed_kernel<<<(unsigned)((n + 255) / 256), 256, 0, stream>>>(
            sentence, pos1, pos2, word_emb, p1t, p2t, embPad);
        long nh = (long)KT_H * NPAD * 32;
        long nx = (long)KT_X * NPAD * 32;
        pack_w_kernel<<<(unsigned)((nh + 255) / 256), 256, 0, stream>>>(W_hh_f, BpF_hh, KT_H, HID);
        pack_w_kernel<<<(unsigned)((nh + 255) / 256), 256, 0, stream>>>(W_hh_b, BpB_hh, KT_H, HID);
        pack_w_kernel<<<(unsigned)((nx + 255) / 256), 256, 0, stream>>>(W_ih_f, BpF_ih, KT_X, D_IN);
        pack_w_kernel<<<(unsigned)((nx + 255) / 256), 256, 0, stream>>>(W_ih_b, BpB_ih, KT_X, D_IN);
    }

    // -------- bidirectional GRU time loop (WMMA GEMMs + gates) --------
    const dim3 gemmGrid(N_TOT / 32, NPAD / 32);   // 64 x 22
    const unsigned gateBlocks = (N_TOT * HID + 255) / 256;
    for (int s = 0; s < SEQ; ++s) {
        const int tf = s, tb = SEQ - 1 - s;
        // forward: gh = h_f @ W_hh_f^T ; gi = x[tf] @ W_ih_f^T
        wmma_gemm_bf16_kernel<<<gemmGrid, 32, 0, stream>>>(hbf_f, KPAD_H, BpF_hh, gh_f, NPAD, KT_H);
        wmma_gemm_bf16_kernel<<<gemmGrid, 32, 0, stream>>>(embPad + (size_t)tf * KPAD_X, SEQ * KPAD_X,
                                                           BpF_ih, gi_f, NPAD, KT_X);
        // backward
        wmma_gemm_bf16_kernel<<<gemmGrid, 32, 0, stream>>>(hbf_b, KPAD_H, BpB_hh, gh_b, NPAD, KT_H);
        wmma_gemm_bf16_kernel<<<gemmGrid, 32, 0, stream>>>(embPad + (size_t)tb * KPAD_X, SEQ * KPAD_X,
                                                           BpB_ih, gi_b, NPAD, KT_X);
        gru_gate_kernel<<<gateBlocks, 256, 0, stream>>>(gh_f, gi_f, b_ih_f, b_hh_f, hf_f, hbf_f, tup, tf);
        gru_gate_kernel<<<gateBlocks, 256, 0, stream>>>(gh_b, gi_b, b_ih_b, b_hh_b, hf_b, hbf_b, tup, tb);
    }

    // -------- attention + loss --------
    word_attn_kernel<<<N_TOT, 256, 0, stream>>>(tup, attn_w, repre);
    bag_attn_kernel<<<NBAG, 256, 0, stream>>>(repre, sen_a, sen_r, tshape, sen_s);
    logits_loss_kernel<<<NBAG, 128, 0, stream>>>(sen_s, rel_emb, sen_d, y_batch, out);
}
